// SparseMLP_33122787787377
// MI455X (gfx1250) — matrix operand order repaired and checked by reading, not compiled, not run on hardware
//
#include <hip/hip_runtime.h>
#include <hip/hip_bf16.h>

// ---------------------------------------------------------------------------
// gpt-oss MoE MLP for MI455X (gfx1250): bf16 WMMA GEMMs with f32 accumulate.
// B=1, S=512, H=2048, E=16, K=4, I=2048.
// ---------------------------------------------------------------------------

typedef __attribute__((ext_vector_type(16))) __bf16 v16bf;
typedef __attribute__((ext_vector_type(8)))  float  v8f;

union U16 { uint4 q[2]; v16bf v; };
union BPK { __bf16 h[2]; unsigned int u; };

#define NTOK   512
#define NH     2048
#define NE     16
#define NK     4
#define NI     2048
#define N2I    4096
#define ALPHA  1.702f
#define LIMIT  7.0f
#define KC     64            // K chunk staged in LDS per iteration

// workspace layout (bytes)
#define WS_COUNTS   0            // 16 * int
#define WS_TOPIDX   4096         // 512*4 int
#define WS_TOPPROB  16384        // 512*4 float
#define WS_SLOTMAP  32768        // 16*512 int  (token | (kslot<<16))
#define WS_ACT      65536        // 16*512*2048 bf16   (33,554,432 B)
#define WS_PAIR     33619968     // 512*4*2048 float   (16,777,216 B)

// ---------------------------------------------------------------------------
// Kernel 1: router.  One block per token.
// ---------------------------------------------------------------------------
__global__ void moe_router_kernel(const float* __restrict__ hs,
                                  const float* __restrict__ rw,
                                  const float* __restrict__ rb,
                                  float* __restrict__ scores_out,
                                  int* __restrict__ counts,
                                  int* __restrict__ top_idx,
                                  float* __restrict__ top_prob,
                                  int* __restrict__ slotmap) {
    const int t = blockIdx.x;
    const int tid = threadIdx.x;          // 256 threads
    __shared__ float red[NE][256];

    float acc[NE];
#pragma unroll
    for (int e = 0; e < NE; ++e) acc[e] = 0.0f;

    const float* x = hs + (size_t)t * NH;
    for (int h = tid; h < NH; h += 256) {
        const float xv = x[h];
        const float* wrow = rw + (size_t)h * NE;
#pragma unroll
        for (int e = 0; e < NE; ++e) acc[e] += xv * wrow[e];
    }
#pragma unroll
    for (int e = 0; e < NE; ++e) red[e][tid] = acc[e];
    __syncthreads();
    for (int s = 128; s > 0; s >>= 1) {
        if (tid < s) {
#pragma unroll
            for (int e = 0; e < NE; ++e) red[e][tid] += red[e][tid + s];
        }
        __syncthreads();
    }

    if (tid == 0) {
        float logits[NE];
#pragma unroll
        for (int e = 0; e < NE; ++e) logits[e] = red[e][0] + rb[e];

        int   idx[NK];
        float val[NK];
        bool  used[NE];
#pragma unroll
        for (int e = 0; e < NE; ++e) used[e] = false;
        for (int k = 0; k < NK; ++k) {
            int best = 0; float bv = -3.0e38f;
            for (int e = 0; e < NE; ++e)
                if (!used[e] && logits[e] > bv) { bv = logits[e]; best = e; }
            used[best] = true; idx[k] = best; val[k] = bv;
        }
        const float m = val[0];
        float p[NK], se = 0.0f;
        for (int k = 0; k < NK; ++k) { p[k] = __expf(val[k] - m); se += p[k]; }
        const float inv = 1.0f / se;

        float sc[NE];
#pragma unroll
        for (int e = 0; e < NE; ++e) sc[e] = 0.0f;
        for (int k = 0; k < NK; ++k) {
            const float pr = p[k] * inv;
            sc[idx[k]] = pr;
            top_idx[t * NK + k]  = idx[k];
            top_prob[t * NK + k] = pr;
            const int slot = atomicAdd(&counts[idx[k]], 1);
            slotmap[idx[k] * NTOK + slot] = t | (k << 16);
        }
#pragma unroll
        for (int e = 0; e < NE; ++e) scores_out[t * NE + e] = sc[e];
    }
}

// ---------------------------------------------------------------------------
// Shared WMMA fragment helpers (ISA 7.12.2 layouts, wave32).
// A tile LDS: row-major [M][KC] bf16 ; lane(m=l16, half=hl) needs
//   k = kk + hl*8 + 0..7  and  k = kk + 16 + hl*8 + 0..7  (two 16B runs)
// B tile LDS: [N][KC] bf16 (transposed) ; lane(n=l16, half=hl) needs
//   k = kk + hl*8 + 0..7  and  k = kk + 16 + hl*8 + 0..7  (two 16B runs)
// ---------------------------------------------------------------------------
__device__ __forceinline__ v16bf load_fragA(const __bf16* rowp, int kk, int hl) {
    U16 a;
    a.q[0] = *reinterpret_cast<const uint4*>(rowp + kk + hl * 8);
    a.q[1] = *reinterpret_cast<const uint4*>(rowp + kk + 16 + hl * 8);
    return a.v;
}
__device__ __forceinline__ v16bf load_fragB(const __bf16* rowp, int kk, int hl) {
    U16 b;
    b.q[0] = *reinterpret_cast<const uint4*>(rowp + kk + hl * 8);
    b.q[1] = *reinterpret_cast<const uint4*>(rowp + kk + 16 + hl * 8);
    return b.v;
}

// ---------------------------------------------------------------------------
// Kernel 2: gate_up GEMM + activation.
// grid = (4096/64 col strips, 8 token groups, 16 experts), block = 256 (8 waves)
// Each block: 64 gathered tokens x 64 gate_up columns, K=2048 in chunks of 64.
// Each wave owns 2 C tiles (16x16 f32). A = tokens (bf16), B = W^T (bf16).
// ---------------------------------------------------------------------------
__global__ void moe_gateup_kernel(const float* __restrict__ hs,
                                  const float* __restrict__ gup,
                                  const float* __restrict__ gub,
                                  const int* __restrict__ counts,
                                  const int* __restrict__ slotmap,
                                  __bf16* __restrict__ act) {
    const int e = blockIdx.z;
    const int count = counts[e];
    const int g0 = blockIdx.y * 64;
    if (g0 >= count) return;
    const int c0 = blockIdx.x * 64;      // base column in 2I dimension

    const int tid  = threadIdx.x;
    const int wave = tid >> 5;
    const int lane = tid & 31;
    const int hl   = lane >> 4;          // which 16-lane half
    const int l16  = lane & 15;

    __shared__ __bf16 Abuf[64][KC];
    __shared__ __bf16 Bbuf[64][KC];
    __shared__ float  gu[64][65];
    __shared__ int    rowtok[64];

    if (tid < 64) {
        const int slot = g0 + tid;
        rowtok[tid] = (slot < count) ? (slotmap[e * NTOK + slot] & 0xffff) : -1;
    }
    __syncthreads();

    const int mt  = wave & 3;            // token tile 0..3
    const int nt0 = (wave >> 2) * 2;     // first of two column tiles

    v8f cacc0 = {};
    v8f cacc1 = {};
    const float* W = gup + (size_t)e * NH * N2I;

    // staging maps (per thread, per chunk)
    const int arow  = tid >> 2;          // A: row 0..63
    const int akoff = (tid & 3) * 16;    // A: 16 consecutive k
    const int atok  = rowtok[arow];
    const int bk    = (tid >> 3) * 2;    // B: k pair base 0..62
    const int bcoff = (tid & 7) * 8;     // B: 8 consecutive columns

    for (int k0 = 0; k0 < NH; k0 += KC) {
        // ---- stage A: 64 rows x KC k  (f32 -> bf16, contiguous dst) ----
        if (atok >= 0) {
            const float* src = hs + (size_t)atok * NH + k0 + akoff;
#pragma unroll
            for (int i = 0; i < 16; ++i) Abuf[arow][akoff + i] = (__bf16)src[i];
        } else {
#pragma unroll
            for (int i = 0; i < 16; ++i) Abuf[arow][akoff + i] = (__bf16)0.0f;
        }
        // ---- stage B transposed: Bbuf[n][k] = W[k0+k][c0+n], packed pairs ----
        {
            const float* s0 = W + (size_t)(k0 + bk) * N2I + c0 + bcoff;
            const float* s1 = s0 + N2I;
#pragma unroll
            for (int i = 0; i < 8; ++i) {
                BPK p;
                p.h[0] = (__bf16)s0[i];
                p.h[1] = (__bf16)s1[i];
                *reinterpret_cast<unsigned int*>(&Bbuf[bcoff + i][bk]) = p.u;
            }
        }
        __syncthreads();

        const __bf16* arowp  = &Abuf[mt * 16 + l16][0];
        const __bf16* brow0p = &Bbuf[nt0 * 16 + l16][0];
        const __bf16* brow1p = &Bbuf[(nt0 + 1) * 16 + l16][0];
#pragma unroll
        for (int kk = 0; kk < KC; kk += 32) {
            const v16bf a  = load_fragA(arowp, kk, hl);
            const v16bf b0 = load_fragB(brow0p, kk, hl);
            const v16bf b1 = load_fragB(brow1p, kk, hl);
            cacc0 = __builtin_amdgcn_wmma_f32_16x16x32_bf16(false, a, false, b0,
                                                            (short)0, cacc0, false, false);
            cacc1 = __builtin_amdgcn_wmma_f32_16x16x32_bf16(false, a, false, b1,
                                                            (short)0, cacc1, false, false);
        }
        __syncthreads();
    }

    // spill C tiles to LDS: gu[token][col]
#pragma unroll
    for (int v = 0; v < 8; ++v) {
        const int m = v + 8 * hl;
        gu[mt * 16 + m][nt0 * 16 + l16]       = cacc0[v];
        gu[mt * 16 + m][(nt0 + 1) * 16 + l16] = cacc1[v];
    }
    __syncthreads();

    // activation: 64 tokens x 32 act columns (pairs of gate_up columns)
    {
        const int row = tid >> 2;
        const int j0  = (tid & 3) * 8;
        const int slot = g0 + row;
        if (slot < count) {
            const float* gb = gub + (size_t)e * N2I + c0;
            __bf16* dst = act + ((size_t)(e * NTOK + slot)) * NI + (c0 >> 1);
#pragma unroll
            for (int i = 0; i < 8; ++i) {
                const int j = j0 + i;
                float g = gu[row][2 * j]     + gb[2 * j];
                float u = gu[row][2 * j + 1] + gb[2 * j + 1];
                g = fminf(g, LIMIT);
                u = fminf(fmaxf(u, -LIMIT), LIMIT);
                const float sig = 1.0f / (1.0f + __expf(-ALPHA * g));
                dst[j] = (__bf16)((u + 1.0f) * (g * sig));
            }
        }
    }
}

// ---------------------------------------------------------------------------
// Kernel 3: down GEMM.  grid = (2048/64 h strips, 8 token groups, 16 experts)
// Output scattered into dense pair buffer [token][kslot][H]  (no float atomics)
// ---------------------------------------------------------------------------
__global__ void moe_down_kernel(const __bf16* __restrict__ act,
                                const float* __restrict__ dwn,
                                const int* __restrict__ counts,
                                const int* __restrict__ slotmap,
                                float* __restrict__ pair_down) {
    const int e = blockIdx.z;
    const int count = counts[e];
    const int g0 = blockIdx.y * 64;
    if (g0 >= count) return;
    const int h0 = blockIdx.x * 64;

    const int tid  = threadIdx.x;
    const int wave = tid >> 5;
    const int lane = tid & 31;
    const int hl   = lane >> 4;
    const int l16  = lane & 15;

    __shared__ __bf16 Abuf[64][KC];
    __shared__ __bf16 Bbuf[64][KC];
    __shared__ int    rowpack[64];

    if (tid < 64) {
        const int slot = g0 + tid;
        rowpack[tid] = (slot < count) ? slotmap[e * NTOK + slot] : -1;
    }
    __syncthreads();

    const int mt  = wave & 3;
    const int nt0 = (wave >> 2) * 2;

    v8f cacc0 = {};
    v8f cacc1 = {};
    const float* W = dwn + (size_t)e * NI * NH;

    const int arow   = tid >> 2;
    const int akoff  = (tid & 3) * 16;
    const bool arowv = (g0 + arow) < count;
    const int bk     = (tid >> 3) * 2;
    const int bhoff  = (tid & 7) * 8;

    for (int k0 = 0; k0 < NI; k0 += KC) {
        // ---- stage A (already bf16): 32B contiguous per thread ----
        if (arowv) {
            const uint4* src = reinterpret_cast<const uint4*>(
                act + ((size_t)(e * NTOK + g0 + arow)) * NI + k0 + akoff);
            uint4* dst = reinterpret_cast<uint4*>(&Abuf[arow][akoff]);
            dst[0] = src[0];
            dst[1] = src[1];
        } else {
#pragma unroll
            for (int i = 0; i < 16; ++i) Abuf[arow][akoff + i] = (__bf16)0.0f;
        }
        // ---- stage B transposed: Bbuf[n][k] = W[k0+k][h0+n], packed pairs ----
        {
            const float* s0 = W + (size_t)(k0 + bk) * NH + h0 + bhoff;
            const float* s1 = s0 + NH;
#pragma unroll
            for (int i = 0; i < 8; ++i) {
                BPK p;
                p.h[0] = (__bf16)s0[i];
                p.h[1] = (__bf16)s1[i];
                *reinterpret_cast<unsigned int*>(&Bbuf[bhoff + i][bk]) = p.u;
            }
        }
        __syncthreads();

        const __bf16* arowp  = &Abuf[mt * 16 + l16][0];
        const __bf16* brow0p = &Bbuf[nt0 * 16 + l16][0];
        const __bf16* brow1p = &Bbuf[(nt0 + 1) * 16 + l16][0];
#pragma unroll
        for (int kk = 0; kk < KC; kk += 32) {
            const v16bf a  = load_fragA(arowp, kk, hl);
            const v16bf b0 = load_fragB(brow0p, kk, hl);
            const v16bf b1 = load_fragB(brow1p, kk, hl);
            cacc0 = __builtin_amdgcn_wmma_f32_16x16x32_bf16(false, a, false, b0,
                                                            (short)0, cacc0, false, false);
            cacc1 = __builtin_amdgcn_wmma_f32_16x16x32_bf16(false, a, false, b1,
                                                            (short)0, cacc1, false, false);
        }
        __syncthreads();
    }

    // scatter D tiles into pair buffer
#pragma unroll
    for (int v = 0; v < 8; ++v) {
        const int m = v + 8 * hl;
        const int slot = g0 + mt * 16 + m;
        if (slot < count) {
            const int packed = rowpack[mt * 16 + m];
            const int t  = packed & 0xffff;
            const int kk = packed >> 16;
            float* dst = pair_down + ((size_t)(t * NK + kk)) * NH + h0;
            dst[nt0 * 16 + l16]       = cacc0[v];
            dst[(nt0 + 1) * 16 + l16] = cacc1[v];
        }
    }
}

// ---------------------------------------------------------------------------
// Kernel 4: weighted combine with down bias (fixed-order deterministic sum).
// ---------------------------------------------------------------------------
__global__ void moe_combine_kernel(const float* __restrict__ pair_down,
                                   const float* __restrict__ dbias,
                                   const int* __restrict__ top_idx,
                                   const float* __restrict__ top_prob,
                                   float* __restrict__ out) {
    const int idx = blockIdx.x * 256 + threadIdx.x;
    if (idx >= NTOK * NH) return;
    const int t = idx >> 11;     // /2048
    const int h = idx & (NH - 1);
    float s = 0.0f;
#pragma unroll
    for (int k = 0; k < NK; ++k) {
        const int e = top_idx[t * NK + k];
        s += top_prob[t * NK + k] *
             (pair_down[((size_t)(t * NK + k)) * NH + h] + dbias[(size_t)e * NH + h]);
    }
    out[idx] = s;
}

// ---------------------------------------------------------------------------
extern "C" void kernel_launch(void* const* d_in, const int* in_sizes, int n_in,
                              void* d_out, int out_size, void* d_ws, size_t ws_size,
                              hipStream_t stream) {
    const float* hs    = (const float*)d_in[0];   // [1,512,2048]
    const float* rw    = (const float*)d_in[1];   // [2048,16]
    const float* rb    = (const float*)d_in[2];   // [16]
    const float* gup   = (const float*)d_in[3];   // [16,2048,4096]
    const float* gub   = (const float*)d_in[4];   // [16,4096]
    const float* dwn   = (const float*)d_in[5];   // [16,2048,2048]
    const float* dbias = (const float*)d_in[6];   // [16,2048]

    float* out_main   = (float*)d_out;                  // [512,2048]
    float* out_scores = out_main + (size_t)NTOK * NH;   // [512,16]

    char* ws = (char*)d_ws;
    int*    counts   = (int*)(ws + WS_COUNTS);
    int*    top_idx  = (int*)(ws + WS_TOPIDX);
    float*  top_prob = (float*)(ws + WS_TOPPROB);
    int*    slotmap  = (int*)(ws + WS_SLOTMAP);
    __bf16* act      = (__bf16*)(ws + WS_ACT);
    float*  pair_dn  = (float*)(ws + WS_PAIR);

    hipMemsetAsync(counts, 0, NE * sizeof(int), stream);

    moe_router_kernel<<<NTOK, 256, 0, stream>>>(hs, rw, rb, out_scores,
                                                counts, top_idx, top_prob, slotmap);

    dim3 gridB(N2I / 64, NTOK / 64, NE);   // 64 x 8 x 16
    moe_gateup_kernel<<<gridB, 256, 0, stream>>>(hs, gup, gub, counts, slotmap, act);

    dim3 gridC(NH / 64, NTOK / 64, NE);    // 32 x 8 x 16
    moe_down_kernel<<<gridC, 256, 0, stream>>>(act, dwn, counts, slotmap, pair_dn);

    moe_combine_kernel<<<(NTOK * NH) / 256, 256, 0, stream>>>(pair_dn, dbias,
                                                              top_idx, top_prob, out_main);
}
